// DANSE_Supervised_86809878986920
// MI455X (gfx1250) — matrix operand order repaired and checked by reading, not compiled, not run on hardware
//
#include <hip/hip_runtime.h>
#include <hip/hip_bf16.h>
#include <math.h>

// ---------------------------------------------------------------------------
// DANSE supervised log-prob, CDNA5 (gfx1250) implementation.
//   Stage 1: persistent GRU (WMMA bf16, TDM-prefetched inputs, bf16 h in LDS)
//   Stage 2: dense head (WMMA bf16, 16-row tiles over all N*T rows)
//   Stage 3: per-(n,t) 10x10 Kalman update + Gaussian logpdf (register Cholesky)
//   Stage 4: scalar finalize
// ---------------------------------------------------------------------------

typedef __bf16 v16bf __attribute__((ext_vector_type(16)));
typedef float  v8f   __attribute__((ext_vector_type(8)));
typedef unsigned int u32x4 __attribute__((ext_vector_type(4)));
typedef int          i32x4 __attribute__((ext_vector_type(4)));
typedef int          i32x8 __attribute__((ext_vector_type(8)));

constexpr int kN = 128, kT = 1000, kNS = 10, kNO = 10, kHID = 64, kDENSE = 32;
constexpr int GPAD  = 65;  // LDS row pitch for r/z gate buffers (f32, 16 x 64)
constexpr int HBPAD = 66;  // LDS row pitch for bf16 h image (16 x 64), 4B-aligned rows
constexpr int YPAD  = 33;  // LDS row pitch for y buffer (f32, 16 x 32)

static __device__ inline v8f wmma_bf16(v16bf a, v16bf b, v8f c) {
  // D = A(16x32) * B(32x16) + C, f32 accumulate
  return __builtin_amdgcn_wmma_f32_16x16x32_bf16(false, a, false, b, (short)0, c, false, false);
}

// A-operand (16x32 bf16) per ISA 7.12.2:
//   lanes 0-15:  row M = lane,    VGPR0-3 hold K=0..7,  VGPR4-7 hold K=16..23
//   lanes 16-31: row M = lane-16, VGPR0-3 hold K=8..15, VGPR4-7 hold K=24..31
static __device__ inline v16bf load_A16x32_f32(const float* base, int rstride, int koff, int lane) {
  int row = lane & 15, hi = lane >> 4;
  const float* rp = base + (size_t)row * rstride + koff;
  v16bf a;
#pragma unroll
  for (int v = 0; v < 8; ++v) {
    int k = ((v < 4) ? 2 * v : 16 + 2 * (v - 4)) + 8 * hi;
    a[2 * v]     = (__bf16)rp[k];
    a[2 * v + 1] = (__bf16)rp[k + 1];
  }
  return a;
}

// Same, but source already bf16 (pairs merge into single b32 loads).
static __device__ inline v16bf load_A16x32_bf16(const __bf16* base, int rstride, int koff, int lane) {
  int row = lane & 15, hi = lane >> 4;
  const __bf16* rp = base + (size_t)row * rstride + koff;
  v16bf a;
#pragma unroll
  for (int v = 0; v < 8; ++v) {
    int k = ((v < 4) ? 2 * v : 16 + 2 * (v - 4)) + 8 * hi;
    a[2 * v]     = rp[k];
    a[2 * v + 1] = rp[k + 1];
  }
  return a;
}

// Zero-padded K (input projection, K valid = 10); rp = row pointer.
static __device__ inline v16bf load_A16x32_pad(const float* rp, int kvalid, int lane) {
  int hi = lane >> 4;
  v16bf a;
#pragma unroll
  for (int v = 0; v < 8; ++v) {
    int k = ((v < 4) ? 2 * v : 16 + 2 * (v - 4)) + 8 * hi;
    a[2 * v]     = (k     < kvalid) ? (__bf16)rp[k]     : (__bf16)0.0f;
    a[2 * v + 1] = (k + 1 < kvalid) ? (__bf16)rp[k + 1] : (__bf16)0.0f;
  }
  return a;
}

// B-operand (32x16 bf16): N column = lane & 15, element e = K in A's table order.
// B[K][N] = W[(c0+N)*rstride + koff + K]  (W stored row-major [out, in]).
static __device__ inline v16bf load_B32x16(const float* W, int rstride, int c0, int koff,
                                           int kvalid, int nvalid, int lane) {
  int n = lane & 15, hi = lane >> 4;
  v16bf b;
#pragma unroll
  for (int e = 0; e < 16; ++e) {
    int k = ((e < 8) ? e : 16 + (e - 8)) + 8 * hi;
    float f = (k < kvalid && n < nvalid) ? W[(size_t)(c0 + n) * rstride + koff + k] : 0.0f;
    b[e] = (__bf16)f;
  }
  return b;
}

// Fast activations: raw v_rcp_f32 / v_exp_f32 (operands already passed through
// bf16 WMMA; IEEE division fixup in the serial loop is wasted latency).
static __device__ inline float fast_sigmoid(float g) {
  return __builtin_amdgcn_rcpf(1.0f + __expf(-g));   // exp->inf, rcp->0: saturates
}
static __device__ inline float fast_tanh(float x) {
  return 2.0f * __builtin_amdgcn_rcpf(1.0f + __expf(-2.0f * x)) - 1.0f;
}

// ---------------------------------------------------------------------------
// Stage 1: GRU.  grid = N/16 blocks, 384 threads (12 waves).
// Wave j owns gate columns [16j, 16j+16): j<4 -> r, 4..7 -> z, 8..11 -> n.
// h lives in LDS as a packed bf16 image (WMMA A feed); each n-lane keeps its
// own f32 h element in registers for the state update.
// The Yi tile for step t+1 is DMA'd into LDS by the Tensor Data Mover
// (double-buffered), waited with s_wait_tensorcnt.
// ---------------------------------------------------------------------------
__global__ void __launch_bounds__(384)
gru_kernel(const float* __restrict__ Yi, const float* __restrict__ W_ih,
           const float* __restrict__ W_hh, const float* __restrict__ b_ih,
           const float* __restrict__ b_hh, __bf16* __restrict__ h_out) {
  __shared__ __bf16 hbf[16 * HBPAD];       // bf16 image of h (16 x 64)
  __shared__ float  rbuf[16 * GPAD];
  __shared__ float  zbuf[16 * GPAD];
  __shared__ float  xbuf[2][16 * kNO];     // TDM double buffer: 16 rows x 10 f32

  const int tid = threadIdx.x, wave = tid >> 5, lane = tid & 31;
  const int n0 = blockIdx.x * 16;
  const int c0 = wave * 16;                // column tile base within [0,192)
  const int col16 = lane & 15;
  const int hi = lane >> 4, row = lane & 15;
  const bool is_n = (wave >= 8);

  // TDM descriptor issue: load 16x10 f32 tile (row stride T*NO) -> LDS buffer.
  auto issue_tdm = [&](int t, int buf) {
    unsigned lds_addr = (unsigned)(size_t)(void*)&xbuf[buf][0];
    unsigned long long ga =
        (unsigned long long)(size_t)(Yi + ((size_t)n0 * kT + (size_t)t) * kNO);
    u32x4 g0;
    g0[0] = 1u;                                         // count=1 valid descriptor
    g0[1] = lds_addr;                                   // lds_addr [63:32]
    g0[2] = (unsigned)(ga & 0xffffffffu);               // global_addr [95:64]
    g0[3] = (unsigned)((ga >> 32) & 0x1ffffffu) | (2u << 30);  // addr[56:32] | type=2
    i32x8 g1;
    g1[0] = (int)(2u << 16);              // workgroup_mask=0 | data_size=2 (4B)
    g1[1] = (int)((unsigned)kNO << 16);   // tensor_dim0 = 10 (elements)
    g1[2] = (int)(16u << 16);             // tensor_dim1 = 16 (rows)
    g1[3] = (int)((unsigned)kNO << 16);   // tile_dim0 = 10
    g1[4] = 16;                           // tile_dim1 = 16, tile_dim2 = 0
    g1[5] = kT * kNO;                     // tensor_dim0_stride = 10000 elements
    g1[6] = 0;
    g1[7] = 0;
    i32x4 gz4; gz4[0] = 0; gz4[1] = 0; gz4[2] = 0; gz4[3] = 0;
    i32x8 gz8;
#pragma unroll
    for (int i = 0; i < 8; ++i) gz8[i] = 0;
    // 6-arg form (clang-23 / therock-10 toolchain)
    __builtin_amdgcn_tensor_load_to_lds(g0, g1, gz4, gz4, gz8, 0);
  };

  for (int i = tid; i < 16 * HBPAD; i += blockDim.x) hbf[i] = (__bf16)0.0f;
  if (wave == 0) {
    issue_tdm(0, 0);
    __builtin_amdgcn_s_wait_tensorcnt(0);
  }
  __syncthreads();

  const float bih = b_ih[c0 + col16];
  const float bhh = b_hh[c0 + col16];
  const v16bf Bx  = load_B32x16(W_ih, kNO,  c0, 0,  kNO, 16, lane);
  const v16bf Bh0 = load_B32x16(W_hh, kHID, c0, 0,  32,  16, lane);
  const v16bf Bh1 = load_B32x16(W_hh, kHID, c0, 32, 32,  16, lane);

  float hreg[8];                           // n-waves: own h elements (m, c)
#pragma unroll
  for (int v = 0; v < 8; ++v) hreg[v] = 0.0f;

  for (int t = 0; t < kT; ++t) {
    const float* xb = &xbuf[t & 1][0];
    v16bf Ax  = load_A16x32_pad(xb + row * kNO, kNO, lane);
    v16bf Ah0 = load_A16x32_bf16(hbf, HBPAD, 0,  lane);
    v16bf Ah1 = load_A16x32_bf16(hbf, HBPAD, 32, lane);

    if (wave == 0 && t + 1 < kT) issue_tdm(t + 1, (t + 1) & 1);  // async prefetch

    v8f cx, ch;
#pragma unroll
    for (int v = 0; v < 8; ++v) { cx[v] = bih; ch[v] = bhh; }
    cx = wmma_bf16(Ax,  Bx,  cx);   // x-projection part (kept separate for n-gate)
    ch = wmma_bf16(Ah0, Bh0, ch);   // recurrent part
    ch = wmma_bf16(Ah1, Bh1, ch);

    if (!is_n) {
      float* buf = (wave < 4) ? rbuf : zbuf;
      const int cb = (wave < 4) ? c0 : (c0 - 64);
#pragma unroll
      for (int v = 0; v < 8; ++v) {
        int m = v + 8 * hi;
        buf[m * GPAD + cb + col16] = fast_sigmoid(cx[v] + ch[v]);
      }
    }
    __syncthreads();
    if (is_n) {
      const int cb = c0 - 128;
#pragma unroll
      for (int v = 0; v < 8; ++v) {
        int m = v + 8 * hi;
        int c = cb + col16;
        float r  = rbuf[m * GPAD + c];
        float z  = zbuf[m * GPAD + c];
        float nv = fast_tanh(cx[v] + r * ch[v]);
        float hn = (1.0f - z) * nv + z * hreg[v];
        hreg[v] = hn;
        __bf16 hb = (__bf16)hn;
        hbf[m * HBPAD + c] = hb;                                  // feed next step's A
        h_out[((size_t)(n0 + m) * kT + t) * kHID + c] = hb;       // feed head kernel
      }
    }
    if (wave == 0) __builtin_amdgcn_s_wait_tensorcnt(0);          // t+1 tile landed
    __syncthreads();
  }
}

// ---------------------------------------------------------------------------
// Stage 2: dense head.  grid = N*T/16 blocks, 64 threads (2 waves).
//   y = relu(h @ W_fc^T + b_fc); mu = y @ W_mean^T + b_mean;
//   vars_raw = y @ W_vars^T + b_vars (softplus applied in stage 3).
// ---------------------------------------------------------------------------
__global__ void __launch_bounds__(64)
head_kernel(const __bf16* __restrict__ hsrc, const float* __restrict__ W_fc,
            const float* __restrict__ b_fc, const float* __restrict__ W_mean,
            const float* __restrict__ b_mean, const float* __restrict__ W_vars,
            const float* __restrict__ b_vars, float* __restrict__ mu_ws,
            float* __restrict__ vr_ws) {
  __shared__ float ybuf[16 * YPAD];
  const int tid = threadIdx.x, wave = tid >> 5, lane = tid & 31;
  const int m0 = blockIdx.x * 16;
  const int hi = lane >> 4, col16 = lane & 15;
  const int c0 = wave * 16;  // y columns [c0, c0+16)

  v16bf Ah0 = load_A16x32_bf16(hsrc + (size_t)m0 * kHID, kHID, 0,  lane);
  v16bf Ah1 = load_A16x32_bf16(hsrc + (size_t)m0 * kHID, kHID, 32, lane);
  v16bf Bf0 = load_B32x16(W_fc, kHID, c0, 0,  32, 16, lane);
  v16bf Bf1 = load_B32x16(W_fc, kHID, c0, 32, 32, 16, lane);

  v8f acc;
  float bf = b_fc[c0 + col16];
#pragma unroll
  for (int v = 0; v < 8; ++v) acc[v] = bf;
  acc = wmma_bf16(Ah0, Bf0, acc);
  acc = wmma_bf16(Ah1, Bf1, acc);

#pragma unroll
  for (int v = 0; v < 8; ++v)
    ybuf[(v + 8 * hi) * YPAD + c0 + col16] = fmaxf(acc[v], 0.0f);
  __syncthreads();

  v16bf Ay = load_A16x32_f32(ybuf, YPAD, 0, lane);  // 16x32, K = DENSE
  const float* Wm = (wave == 0) ? W_mean : W_vars;
  const float* bm = (wave == 0) ? b_mean : b_vars;
  v16bf Bm = load_B32x16(Wm, kDENSE, 0, 0, 32, kNS, lane);

  v8f c2;
  float bb = (col16 < kNS) ? bm[col16] : 0.0f;
#pragma unroll
  for (int v = 0; v < 8; ++v) c2[v] = bb;
  c2 = wmma_bf16(Ay, Bm, c2);

  float* dst = (wave == 0) ? mu_ws : vr_ws;
  if (col16 < kNS) {
#pragma unroll
    for (int v = 0; v < 8; ++v)
      dst[(size_t)(m0 + v + 8 * hi) * kNS + col16] = c2[v];
  }
}

// ---------------------------------------------------------------------------
// Stage 3: batched Kalman update + Gaussian logpdf, one thread per (n,t).
// SPD structure: L_post = diag(v) - diag(v) H^T S^-1 H diag(v); Cholesky only.
// Accumulates sum over (n,t) of (logdet(L_post) + quad) into *accum.
// ---------------------------------------------------------------------------
static __device__ inline void chol10(float A[kNS][kNS]) {
#pragma unroll
  for (int j = 0; j < 10; ++j) {
    float d = A[j][j];
#pragma unroll
    for (int k = 0; k < j; ++k) d -= A[j][k] * A[j][k];
    d = __builtin_amdgcn_sqrtf(fmaxf(d, 1e-30f));
    A[j][j] = d;
    float inv = __builtin_amdgcn_rcpf(d);
#pragma unroll
    for (int i = j + 1; i < 10; ++i) {
      float s = A[i][j];
#pragma unroll
      for (int k = 0; k < j; ++k) s -= A[i][k] * A[j][k];
      A[i][j] = s * inv;
    }
  }
}

__global__ void __launch_bounds__(256)
kalman_kernel(const float* __restrict__ Yi, const float* __restrict__ Xi,
              const float* __restrict__ Hm, const float* __restrict__ Cw,
              const float* __restrict__ mu_ws, const float* __restrict__ vr_ws,
              float* __restrict__ accum) {
  __shared__ float Hs[kNO * kNS];
  __shared__ float Cs[kNO * kNO];
  const int tid = threadIdx.x;
  if (tid < kNO * kNS) Hs[tid] = Hm[tid];
  if (tid < kNO * kNO) Cs[tid] = Cw[tid];
  __syncthreads();

  const int idx = blockIdx.x * 256 + tid;  // flat (n*T + t)
  float local = 0.0f;
  if (idx < kN * kT) {
    const float* mp = mu_ws + (size_t)idx * kNS;
    const float* vp = vr_ws + (size_t)idx * kNS;
    const float* yo = Yi + (size_t)idx * kNO;
    const float* xo = Xi + (size_t)idx * kNS;

    float mu[kNS], v[kNS];
#pragma unroll
    for (int s = 0; s < kNS; ++s) {
      mu[s] = mp[s];
      float x = vp[s];
      v[s] = (x > 20.0f) ? x : __logf(1.0f + __expf(x));  // softplus
    }

    float innov[kNO];
#pragma unroll
    for (int o = 0; o < kNO; ++o) {
      float s = yo[o];
#pragma unroll
      for (int q = 0; q < kNS; ++q) s -= Hs[o * kNS + q] * mu[q];
      innov[o] = s;
    }

    // S = H diag(v) H^T + C_w  (lower triangle), then Cholesky in place.
    float S[kNO][kNO];
#pragma unroll
    for (int o = 0; o < kNO; ++o)
#pragma unroll
      for (int p = 0; p <= o; ++p) {
        float s = Cs[o * kNO + p];
#pragma unroll
        for (int q = 0; q < kNS; ++q) s += Hs[o * kNS + q] * v[q] * Hs[p * kNS + q];
        S[o][p] = s;
      }
    chol10(S);

    // U = Ls^{-1} H (forward solves; M = U^T U)
    float U[kNO][kNS];
#pragma unroll
    for (int o = 0; o < kNO; ++o) {
      float inv = __builtin_amdgcn_rcpf(S[o][o]);
#pragma unroll
      for (int a = 0; a < kNS; ++a) {
        float s = Hs[o * kNS + a];
#pragma unroll
        for (int j = 0; j < o; ++j) s -= S[o][j] * U[j][a];
        U[o][a] = s * inv;
      }
    }

    // t2 = S^{-1} innov  (forward then backward solve)
    float t1[kNO], t2[kNO];
#pragma unroll
    for (int i = 0; i < kNO; ++i) {
      float s = innov[i];
#pragma unroll
      for (int j = 0; j < i; ++j) s -= S[i][j] * t1[j];
      t1[i] = s * __builtin_amdgcn_rcpf(S[i][i]);
    }
#pragma unroll
    for (int i = kNO - 1; i >= 0; --i) {
      float s = t1[i];
#pragma unroll
      for (int j = i + 1; j < kNO; ++j) s -= S[j][i] * t2[j];
      t2[i] = s * __builtin_amdgcn_rcpf(S[i][i]);
    }

    // mu_post = mu + v .* (H^T t2);  L_post = diag(v) - (v v^T) .* (U^T U)
    float mup[kNS];
#pragma unroll
    for (int s = 0; s < kNS; ++s) {
      float a = 0.0f;
#pragma unroll
      for (int o = 0; o < kNO; ++o) a += Hs[o * kNS + s] * t2[o];
      mup[s] = mu[s] + v[s] * a;
    }

    float Lp[kNS][kNS];
#pragma unroll
    for (int s = 0; s < kNS; ++s)
#pragma unroll
      for (int q = 0; q <= s; ++q) {
        float m = 0.0f;
#pragma unroll
        for (int o = 0; o < kNO; ++o) m += U[o][s] * U[o][q];
        Lp[s][q] = ((s == q) ? v[s] : 0.0f) - v[s] * v[q] * m;
      }
    chol10(Lp);

    float logdet = 0.0f;
#pragma unroll
    for (int s = 0; s < kNS; ++s) logdet += __logf(Lp[s][s]);
    logdet *= 2.0f;

    float zv[kNS], quad = 0.0f;
#pragma unroll
    for (int i = 0; i < kNS; ++i) {
      float s = xo[i] - mup[i];
#pragma unroll
      for (int j = 0; j < i; ++j) s -= Lp[i][j] * zv[j];
      zv[i] = s * __builtin_amdgcn_rcpf(Lp[i][i]);
      quad += zv[i] * zv[i];
    }
    local = logdet + quad;
  }

  __shared__ float red[256];
  red[tid] = local;
  __syncthreads();
#pragma unroll
  for (int s = 128; s > 0; s >>= 1) {
    if (tid < s) red[tid] += red[tid + s];
    __syncthreads();
  }
  if (tid == 0) atomicAdd(accum, red[0]);
}

__global__ void zero_kernel(float* accum) { accum[0] = 0.0f; }

__global__ void finalize_kernel(const float* __restrict__ accum, float* __restrict__ out) {
  if (threadIdx.x == 0) {
    const float c = 0.5f * (float)(kNS * kT) * logf(2.0f * 3.14159265358979323846f);
    out[0] = c - 0.5f * accum[0] / (float)kN;
  }
}

// ---------------------------------------------------------------------------
extern "C" void kernel_launch(void* const* d_in, const int* in_sizes, int n_in,
                              void* d_out, int out_size, void* d_ws, size_t ws_size,
                              hipStream_t stream) {
  const float* Yi     = (const float*)d_in[0];
  const float* Xi     = (const float*)d_in[1];
  const float* H      = (const float*)d_in[2];
  // d_in[3] = mu_w (unused by the reference math)
  const float* Cw     = (const float*)d_in[4];
  const float* W_ih   = (const float*)d_in[5];
  const float* W_hh   = (const float*)d_in[6];
  const float* b_ih   = (const float*)d_in[7];
  const float* b_hh   = (const float*)d_in[8];
  const float* W_fc   = (const float*)d_in[9];
  const float* b_fc   = (const float*)d_in[10];
  const float* W_mean = (const float*)d_in[11];
  const float* b_mean = (const float*)d_in[12];
  const float* W_vars = (const float*)d_in[13];
  const float* b_vars = (const float*)d_in[14];

  char* ws = (char*)d_ws;
  float*  accum = (float*)ws;                                     // 4 B (pad to 256)
  __bf16* h_out = (__bf16*)(ws + 256);                            // N*T*HID bf16 = 16 MB
  float*  mu_ws = (float*)(ws + 256 + (size_t)kN * kT * kHID * 2);
  float*  vr_ws = (float*)((char*)mu_ws + (size_t)kN * kT * kNS * 4);

  hipLaunchKernelGGL(zero_kernel, dim3(1), dim3(1), 0, stream, accum);
  hipLaunchKernelGGL(gru_kernel, dim3(kN / 16), dim3(384), 0, stream,
                     Yi, W_ih, W_hh, b_ih, b_hh, h_out);
  hipLaunchKernelGGL(head_kernel, dim3(kN * kT / 16), dim3(64), 0, stream,
                     h_out, W_fc, b_fc, W_mean, b_mean, W_vars, b_vars, mu_ws, vr_ws);
  hipLaunchKernelGGL(kalman_kernel, dim3(kN * kT / 256), dim3(256), 0, stream,
                     Yi, Xi, H, Cw, mu_ws, vr_ws, accum);
  hipLaunchKernelGGL(finalize_kernel, dim3(1), dim3(32), 0, stream, accum, (float*)d_out);
}
